// Mamba_3410204033867
// MI455X (gfx1250) — matrix operand order repaired and checked by reading, not compiled, not run on hardware
//
#include <hip/hip_runtime.h>
#include <hip/hip_bf16.h>

typedef unsigned short ushort_t;
typedef __attribute__((ext_vector_type(16))) __bf16 v16bf;
typedef __attribute__((ext_vector_type(8)))  __bf16 v8bf;
typedef __attribute__((ext_vector_type(8)))  float  v8f;

// ---------------- constants (match reference) ----------------
#define VV 50257
#define HH 768
#define DD 1536
#define NN 16
#define KK 4
#define RR 48
#define LL 2
#define BB 2
#define TT 1024
#define MM (BB*TT)          // 2048 rows
#define P2 (RR + 2*NN)      // 80
#define KDT 64              // dt GEMM K padded 48 -> 64

__device__ __forceinline__ ushort_t f2bf(float f) {
  unsigned u = __float_as_uint(f);
  unsigned r = u + 0x7fffu + ((u >> 16) & 1u);   // round-to-nearest-even
  return (ushort_t)(r >> 16);
}

// ---------------- embedding gather ----------------
__global__ void k_embed_gather(const int* __restrict__ seq,
                               const float* __restrict__ embed,
                               float* __restrict__ h, long long total) {
  long long i = (long long)blockIdx.x * blockDim.x + threadIdx.x;
  if (i >= total) return;
  int r = (int)(i / HH), c = (int)(i % HH);
  h[i] = embed[(size_t)seq[r] * HH + c];
}

// ---------------- f32 -> bf16 with optional column pad ----------------
__global__ void k_cvt_bf16(const float* __restrict__ src, int sld, int soff,
                           ushort_t* __restrict__ dst, int dld,
                           int cols, long long total) {
  long long i = (long long)blockIdx.x * blockDim.x + threadIdx.x;
  if (i >= total) return;
  int r = (int)(i / dld), c = (int)(i % dld);
  float v = (c < cols) ? src[(size_t)r * sld + soff + c] : 0.0f;
  dst[i] = f2bf(v);
}

// ---------------- RMSNorm (one block per row), bf16 out ----------------
__global__ __launch_bounds__(256) void k_rmsnorm(const float* __restrict__ x,
                                                 const float* __restrict__ w,
                                                 ushort_t* __restrict__ out) {
  __shared__ float red[256];
  int row = blockIdx.x, tid = threadIdx.x;
  const float* xr = x + (size_t)row * HH;
  float s = 0.f;
  for (int c = tid; c < HH; c += 256) { float v = xr[c]; s += v * v; }
  red[tid] = s; __syncthreads();
  for (int st = 128; st > 0; st >>= 1) {
    if (tid < st) red[tid] += red[tid + st];
    __syncthreads();
  }
  float scale = rsqrtf(red[0] / (float)HH + 1e-5f);
  ushort_t* orow = out + (size_t)row * HH;
  for (int c = tid; c < HH; c += 256) orow[c] = f2bf(xr[c] * scale * w[c]);
}

// ---------------- bf16 WMMA GEMM: C[M,N] = A[M,K] * W[N,K]^T (+resid) ----
// one wave per 32x64 output tile; fragments loaded directly per ISA layout
__global__ __launch_bounds__(32) void k_gemm_bf16(
    const ushort_t* __restrict__ A, int lda,
    const ushort_t* __restrict__ Bw, int ldb,
    float* __restrict__ C, int ldc,
    const float* __restrict__ resid, int ldr,
    int N, int Kdim) {
  const int lane = threadIdx.x & 31;
  const int m0 = blockIdx.y * 32;
  const int n0 = blockIdx.x * 64;

  const int am  = lane & 15;
  const int akq = (lane >> 4) << 3;   // A lane chunk base: 0 / 8 halfwords
  const int bn  = lane & 15;
  const int bkq = (lane >> 4) << 3;   // B lane chunk base: 0 / 8 halfwords

  const ushort_t* aRow[2];
#pragma unroll
  for (int i = 0; i < 2; ++i)
    aRow[i] = A + (size_t)(m0 + 16 * i + am) * lda + akq;
  const ushort_t* bRow[4];
#pragma unroll
  for (int j = 0; j < 4; ++j) {
    int nn = n0 + 16 * j + bn; if (nn >= N) nn = N - 1;   // clamp (stores masked)
    bRow[j] = Bw + (size_t)nn * ldb + bkq;
  }

  v8f acc[2][4];
#pragma unroll
  for (int i = 0; i < 2; ++i)
#pragma unroll
    for (int j = 0; j < 4; ++j) acc[i][j] = (v8f)(0.0f);

  for (int k = 0; k < Kdim; k += 32) {
    if (k + 32 < Kdim) {                       // L2 prefetch of next K slab
      __builtin_prefetch(aRow[0] + k + 32, 0, 1);
      __builtin_prefetch(bRow[0] + k + 32, 0, 1);
      __builtin_prefetch(bRow[2] + k + 32, 0, 1);
    }
    v16bf af[2];
#pragma unroll
    for (int i = 0; i < 2; ++i) {
      const v8bf* p = (const v8bf*)(aRow[i] + k);
      v8bf c0 = p[0], c1 = p[2];               // K chunks at +0 and +16 halfwords
      v16bf f;
#pragma unroll
      for (int e = 0; e < 8; ++e) { f[e] = c0[e]; f[e + 8] = c1[e]; }
      af[i] = f;
    }
    v16bf bfm[4];
#pragma unroll
    for (int j = 0; j < 4; ++j) {
      const v8bf* p = (const v8bf*)(bRow[j] + k);
      v8bf c0 = p[0], c1 = p[2];               // K chunks at +0 and +16 halfwords
      v16bf f;
#pragma unroll
      for (int e = 0; e < 8; ++e) { f[e] = c0[e]; f[e + 8] = c1[e]; }
      bfm[j] = f;
    }
#pragma unroll
    for (int i = 0; i < 2; ++i)
#pragma unroll
      for (int j = 0; j < 4; ++j)
        acc[i][j] = __builtin_amdgcn_wmma_f32_16x16x32_bf16(
            false, af[i], false, bfm[j], (short)0, acc[i][j], false, false);
  }

  const int ccol  = lane & 15;
  const int rbase = (lane >> 4) << 3;
#pragma unroll
  for (int i = 0; i < 2; ++i)
#pragma unroll
    for (int j = 0; j < 4; ++j) {
      int col = n0 + 16 * j + ccol;
      if (col < N) {
#pragma unroll
        for (int r = 0; r < 8; ++r) {
          int row = m0 + 16 * i + rbase + r;
          float v = acc[i][j][r];
          if (resid) v += resid[(size_t)row * ldr + col];
          C[(size_t)row * ldc + col] = v;
        }
      }
    }
}

// ---------------- causal depthwise conv (K=4) + SiLU ----------------
__global__ void k_conv_silu(const float* __restrict__ proj,
                            const float* __restrict__ cw,
                            const float* __restrict__ cb,
                            float* __restrict__ xc,
                            ushort_t* __restrict__ xcb, long long total) {
  long long i = (long long)blockIdx.x * blockDim.x + threadIdx.x;
  if (i >= total) return;
  int row = (int)(i / DD), d = (int)(i % DD);
  int b = row / TT, t = row % TT;
  float acc = cb[d];
#pragma unroll
  for (int k = 0; k < KK; ++k) {
    int tt = t - (KK - 1) + k;
    if (tt >= 0)
      acc += cw[d * KK + k] * proj[(size_t)(b * TT + tt) * (2 * DD) + d];
  }
  float s = acc / (1.0f + __expf(-acc));
  xc[i] = s;
  xcb[i] = f2bf(s);
}

// ---------------- dt = softplus(dt_raw + bias), in place ----------------
__global__ void k_softplus(float* __restrict__ dt,
                           const float* __restrict__ dpb, long long total) {
  long long i = (long long)blockIdx.x * blockDim.x + threadIdx.x;
  if (i >= total) return;
  float x = dt[i] + dpb[i % DD];
  dt[i] = (x > 20.f) ? x : log1pf(__expf(x));
}

// ---------------- selective scan: lane per (b,d,n), 16-lane reduce ------
__global__ __launch_bounds__(256) void k_scan(
    const float* __restrict__ dt, const float* __restrict__ xc,
    const float* __restrict__ proj2, const float* __restrict__ proj,
    const float* __restrict__ alog, const float* __restrict__ dpar,
    ushort_t* __restrict__ yb) {
  int idx = blockIdx.x * blockDim.x + threadIdx.x;
  if (idx >= BB * DD * NN) return;
  int n = idx & (NN - 1);
  int p = idx >> 4;            // (b,d) pair index
  int b = p / DD, d = p % DD;

  float Aval = -__expf(alog[d * NN + n]);
  float ddp  = dpar[d];
  float hst  = 0.f;

  for (int t = 0; t < TT; ++t) {
    int row = b * TT + t;
    float sdt = dt[(size_t)row * DD + d];
    float sxc = xc[(size_t)row * DD + d];
    float Bv  = proj2[(size_t)row * P2 + RR + n];
    float Cv  = proj2[(size_t)row * P2 + RR + NN + n];
    hst = __expf(sdt * Aval) * hst + (sdt * sxc) * Bv;
    float c = hst * Cv;
    c += __shfl_xor(c, 1, 32);
    c += __shfl_xor(c, 2, 32);
    c += __shfl_xor(c, 4, 32);
    c += __shfl_xor(c, 8, 32);
    if (n == 0) {
      float g  = proj[(size_t)row * (2 * DD) + DD + d];
      float yv = (c + sxc * ddp) * (g / (1.0f + __expf(-g)));
      yb[(size_t)row * DD + d] = f2bf(yv);
    }
  }
}

// ---------------------------------------------------------------------
extern "C" void kernel_launch(void* const* d_in, const int* in_sizes, int n_in,
                              void* d_out, int out_size, void* d_ws, size_t ws_size,
                              hipStream_t stream) {
  const int*   seq   = (const int*)d_in[0];
  const float* embed = (const float*)d_in[1];
  const float* nwAll = (const float*)d_in[2];
  const float* ipwA  = (const float*)d_in[3];
  const float* cwA   = (const float*)d_in[4];
  const float* cbA   = (const float*)d_in[5];
  const float* xpwA  = (const float*)d_in[6];
  const float* dpwA  = (const float*)d_in[7];
  const float* dpbA  = (const float*)d_in[8];
  const float* alogA = (const float*)d_in[9];
  const float* dparA = (const float*)d_in[10];
  const float* opwA  = (const float*)d_in[11];
  const float* fnw   = (const float*)d_in[12];
  float* out = (float*)d_out;

  // ---- workspace carve-out (aligned 256B) ----
  char* ws = (char*)d_ws;
  size_t off = 0;
  auto carve = [&](size_t bytes) -> char* {
    char* p = ws + off;
    off = (off + bytes + 255) & ~(size_t)255;
    return p;
  };
  float*    hA    = (float*)   carve((size_t)MM * HH * 4);
  float*    hB    = (float*)   carve((size_t)MM * HH * 4);
  ushort_t* xn    = (ushort_t*)carve((size_t)MM * HH * 2);
  float*    proj  = (float*)   carve((size_t)MM * 2 * DD * 4);
  float*    xc    = (float*)   carve((size_t)MM * DD * 4);
  ushort_t* xcb   = (ushort_t*)carve((size_t)MM * DD * 2);
  float*    proj2 = (float*)   carve((size_t)MM * P2 * 4);
  ushort_t* dtin  = (ushort_t*)carve((size_t)MM * KDT * 2);
  float*    dtb   = (float*)   carve((size_t)MM * DD * 4);
  ushort_t* yb    = (ushort_t*)carve((size_t)MM * DD * 2);
  ushort_t* wip   = (ushort_t*)carve((size_t)2 * DD * HH * 2);
  ushort_t* wxp   = (ushort_t*)carve((size_t)P2 * DD * 2);
  ushort_t* wdt   = (ushort_t*)carve((size_t)DD * KDT * 2);
  ushort_t* wop   = (ushort_t*)carve((size_t)HH * DD * 2);
  ushort_t* hfb   = (ushort_t*)carve((size_t)MM * HH * 2);
  ushort_t* emb   = (ushort_t*)carve((size_t)VV * HH * 2);
  (void)ws_size; (void)in_sizes; (void)n_in; (void)out_size;

  auto cvt = [&](const float* src, int sld, int soff, ushort_t* dst, int dld,
                 int rows, int cols) {
    long long total = (long long)rows * dld;
    int blocks = (int)((total + 255) / 256);
    k_cvt_bf16<<<blocks, 256, 0, stream>>>(src, sld, soff, dst, dld, cols, total);
  };
  auto gemm = [&](const ushort_t* A, int lda, const ushort_t* Bw, int ldb,
                  float* C, int ldc, const float* resid, int ldr,
                  int Ncols, int Kdim) {
    dim3 grid((Ncols + 63) / 64, MM / 32);
    k_gemm_bf16<<<grid, 32, 0, stream>>>(A, lda, Bw, ldb, C, ldc, resid, ldr,
                                         Ncols, Kdim);
  };

  // embedding gather + bf16 copy of tied embedding (for logits GEMM)
  {
    long long tot = (long long)MM * HH;
    k_embed_gather<<<(int)((tot + 255) / 256), 256, 0, stream>>>(seq, embed, hA, tot);
    cvt(embed, HH, 0, emb, HH, VV, HH);
  }

  float* hcur = hA;
  float* hnext = hB;
  for (int l = 0; l < LL; ++l) {
    const float* nw_l   = nwAll + (size_t)l * HH;
    const float* ipw_l  = ipwA  + (size_t)l * 2 * DD * HH;
    const float* cw_l   = cwA   + (size_t)l * DD * KK;
    const float* cb_l   = cbA   + (size_t)l * DD;
    const float* xpw_l  = xpwA  + (size_t)l * P2 * DD;
    const float* dpw_l  = dpwA  + (size_t)l * DD * RR;
    const float* dpb_l  = dpbA  + (size_t)l * DD;
    const float* alog_l = alogA + (size_t)l * DD * NN;
    const float* dpar_l = dparA + (size_t)l * DD;
    const float* opw_l  = opwA  + (size_t)l * HH * DD;

    // per-layer weight conversion to bf16 (dt_proj padded K 48->64)
    cvt(ipw_l, HH, 0, wip, HH, 2 * DD, HH);
    cvt(xpw_l, DD, 0, wxp, DD, P2, DD);
    cvt(dpw_l, RR, 0, wdt, KDT, DD, RR);
    cvt(opw_l, DD, 0, wop, DD, HH, DD);

    k_rmsnorm<<<MM, 256, 0, stream>>>(hcur, nw_l, xn);
    gemm(xn, HH, wip, HH, proj, 2 * DD, nullptr, 0, 2 * DD, HH);      // in_proj

    {
      long long tot = (long long)MM * DD;
      k_conv_silu<<<(int)((tot + 255) / 256), 256, 0, stream>>>(
          proj, cw_l, cb_l, xc, xcb, tot);
    }
    gemm(xcb, DD, wxp, DD, proj2, P2, nullptr, 0, P2, DD);            // x_proj
    cvt(proj2, P2, 0, dtin, KDT, MM, RR);                             // dt_in pad
    gemm(dtin, KDT, wdt, KDT, dtb, DD, nullptr, 0, DD, KDT);          // dt_proj
    {
      long long tot = (long long)MM * DD;
      k_softplus<<<(int)((tot + 255) / 256), 256, 0, stream>>>(dtb, dpb_l, tot);
    }
    k_scan<<<(BB * DD * NN + 255) / 256, 256, 0, stream>>>(
        dtb, xc, proj2, proj, alog_l, dpar_l, yb);
    gemm(yb, DD, wop, DD, hnext, HH, hcur, HH, HH, DD);               // out_proj+res

    float* tmp = hcur; hcur = hnext; hnext = tmp;
  }

  k_rmsnorm<<<MM, 256, 0, stream>>>(hcur, fnw, hfb);
  gemm(hfb, HH, emb, HH, out, VV, nullptr, 0, VV, HH);                // logits
}
